// MultiHeadGCNGATMergeVGAE_86414741995585
// MI455X (gfx1250) — matrix operand order repaired and checked by reading, not compiled, not run on hardware
//
#include <hip/hip_runtime.h>
#include <hip/hip_bf16.h>
#include <math.h>

#define N_NODES 8192
#define F_DIM   512
#define H1D     128
#define H2D     64
#define HEADS   4
#define E_EDGES 262144
#define CATW    (2*H1D + 2*HEADS*H1D)   // 1280

#define BM 128
#define BN 128
#define BK 32
#define ASTRIDE 36   // f32 elems per A row in LDS (16B aligned, conflict-free)
#define BSTRIDE 40   // f16 elems per B row in LDS (8B aligned, conflict-free)

typedef __attribute__((ext_vector_type(16))) _Float16 v16h;
typedef __attribute__((ext_vector_type(8)))  float    v8f;

union Frag16 { v16h v; _Float16 h[16]; };

// ---------------- CDNA5 async global->LDS (ASYNCcnt-tracked, ISA ch10.7) ----------------

__device__ __forceinline__ void async_copy_b128(const float* gsrc, void* ldst) {
    unsigned l = (unsigned)(uintptr_t)ldst;   // low 32 bits of flat shared addr = LDS offset
    asm volatile("global_load_async_to_lds_b128 %0, %1, off"
                 :: "v"(l), "v"(gsrc)
                 : "memory");
}
__device__ __forceinline__ void async_wait0() {
    asm volatile("s_wait_asynccnt 0" ::: "memory");
}

// ---------------- helpers ----------------

__device__ __forceinline__ void edge_sd(int e, const int* __restrict__ src,
                                        const int* __restrict__ dst, int E,
                                        int& s, int& d) {
    if (e < E) { s = src[e]; d = dst[e]; }
    else       { s = e - E; d = e - E; }      // self loops appended
}

__device__ __forceinline__ float atomicMaxF(float* addr, float val) {
    int* ai = (int*)addr;
    int old = __float_as_int(*addr);
    while (__int_as_float(old) < val) {
        int assumed = old;
        old = atomicCAS(ai, assumed, __float_as_int(val));
        if (old == assumed) break;
    }
    return __int_as_float(old);
}

// ---------------- tiny utility kernels ----------------

__global__ void k_fill(float* p, float v, int n) {
    int i = blockIdx.x * blockDim.x + threadIdx.x;
    if (i < n) p[i] = v;
}

__global__ void k_init_bias(float* out, const float* __restrict__ bias,
                            int ncols, int ldo, int n) {
    int i = blockIdx.x * blockDim.x + threadIdx.x;
    int total = n * ncols;
    if (i >= total) return;
    int r = i / ncols, c = i - r * ncols;
    out[(size_t)r * ldo + c] = bias ? bias[c] : 0.f;
}

__global__ void k_deg(const int* __restrict__ dst, float* deg, int E) {
    int e = blockIdx.x * blockDim.x + threadIdx.x;
    if (e < E) atomicAdd(&deg[dst[e]], 1.f);
}

__global__ void k_rsqrt(float* p, int n) {
    int i = blockIdx.x * blockDim.x + threadIdx.x;
    if (i < n) p[i] = rsqrtf(p[i]);
}

// ---------------- LDS-tiled WMMA GEMM ----------------
// C[M,N] = act(A[M,K] @ op(B) + bias).  Block = 256 threads = 8 waves
// computing a 128x128 tile; each wave owns a 32x64 sub-tile = 2x4 WMMA tiles.
// A tile staged f32 via async global->LDS; B tile staged f16 transposed [n][k].
// M must be a multiple of 128; K a multiple of 32; N arbitrary multiple of 16.
__global__ __launch_bounds__(256)
void k_gemm_tiled(const float* __restrict__ A, int lda,
                  const float* __restrict__ B, int ldb, int transB,
                  float* __restrict__ C, int ldc,
                  const float* __restrict__ bias,
                  int M, int N, int K, int sigmoid_act) {
    __shared__ float    As[BM * ASTRIDE];   // 18 KB
    __shared__ _Float16 Bs[BN * BSTRIDE];   // 10 KB

    const int t    = threadIdx.x;
    const int wave = t >> 5;
    const int lane = t & 31;
    const int wm   = wave >> 1;             // 0..3  (M direction, 32 rows each)
    const int wn   = wave & 1;              // 0..1  (N direction, 64 cols each)
    const int l15  = lane & 15;
    const int hi   = lane >> 4;             // 0 | 1
    const int bm   = blockIdx.y, bn = blockIdx.x;
    const int rowbase = bm * BM;

    v8f acc[2][4] = {};

    for (int k0 = 0; k0 < K; k0 += BK) {
        // ---- stage A tile [128 x 32] f32, async DMA to LDS ----
#pragma unroll
        for (int i = 0; i < 4; i++) {
            int c  = t + 256 * i;           // 1024 chunks of 16B
            int r  = c >> 3;                // 8 chunks per row
            int kc = (c & 7) << 2;          // 4 floats per chunk
            async_copy_b128(A + (size_t)(rowbase + r) * lda + k0 + kc,
                            &As[r * ASTRIDE + kc]);
        }
        // ---- stage B tile, f32->f16 transpose into Bs[n][k] ----
        if (transB) {                       // B operand = Bt[N,K]; contiguous in k
#pragma unroll
            for (int i = 0; i < 16; i++) {
                int idx = t + 256 * i;
                int n = idx >> 5, k = idx & 31;
                int col = bn * BN + n;
                float v = (col < N) ? B[(size_t)col * ldb + k0 + k] : 0.f;
                Bs[n * BSTRIDE + k] = (_Float16)v;
            }
        } else {                            // B[K,N]; contiguous in n
#pragma unroll
            for (int i = 0; i < 16; i++) {
                int idx = t + 256 * i;
                int k = idx >> 7, n = idx & 127;
                int col = bn * BN + n;
                __builtin_prefetch(&B[(size_t)(k0 + k + BK) * ldb + col], 0, 1);
                float v = (col < N) ? B[(size_t)(k0 + k) * ldb + col] : 0.f;
                Bs[n * BSTRIDE + k] = (_Float16)v;
            }
        }
        async_wait0();
        __syncthreads();

        // ---- load fragments from LDS (ISA 7.12.2 layouts) ----
        Frag16 af[2], bf[4];
        const int kb  = hi ? 8 : 0;         // A: lanes16-31 hold K 8..15 / 24..31
        const int kb2 = hi ? 16 : 0;        // unused: B loads use kb (lanes16-31 hold K 8..15 / 24..31, like A)
#pragma unroll
        for (int i = 0; i < 2; i++) {
            int row = wm * 32 + i * 16 + l15;
            const float* ap = &As[row * ASTRIDE + kb];
#pragma unroll
            for (int x = 0; x < 8; x++) af[i].h[x]     = (_Float16)ap[x];
#pragma unroll
            for (int x = 0; x < 8; x++) af[i].h[8 + x] = (_Float16)ap[16 + x];
        }
#pragma unroll
        for (int j = 0; j < 4; j++) {
            int col = wn * 64 + j * 16 + l15;
            const _Float16* bp = &Bs[col * BSTRIDE + kb];
#pragma unroll
            for (int x = 0; x < 16; x++) bf[j].h[x] = bp[(x < 8) ? x : x + 8];
        }
        // ---- 2x4 WMMA tiles ----
#pragma unroll
        for (int i = 0; i < 2; i++)
#pragma unroll
            for (int j = 0; j < 4; j++)
                acc[i][j] = __builtin_amdgcn_wmma_f32_16x16x32_f16(
                    false, af[i].v, false, bf[j].v,
                    (short)0, acc[i][j], false, false);
        __syncthreads();
    }

    // ---- epilogue: D layout VGPR v -> M = v + 8*hi, N = l15 ----
#pragma unroll
    for (int i = 0; i < 2; i++)
#pragma unroll
        for (int j = 0; j < 4; j++) {
            int ncol = bn * BN + wn * 64 + j * 16 + l15;
            if (ncol < N) {
                int m0 = rowbase + wm * 32 + i * 16 + hi * 8;
                float bb = bias ? bias[ncol] : 0.f;
#pragma unroll
                for (int v = 0; v < 8; v++) {
                    float val = acc[i][j][v] + bb;
                    if (sigmoid_act) val = 1.f / (1.f + __expf(-val));
                    C[(size_t)(m0 + v) * ldc + ncol] = val;
                }
            }
        }
}

// ---------------- GCN aggregation: out[d] += h[s] * dinv[s]*dinv[d] ----------------
__global__ void k_gcn_agg(const float* __restrict__ h, int ldh,
                          const int* __restrict__ src, const int* __restrict__ dst,
                          int E, int n, const float* __restrict__ dinv,
                          float* out, int ldo, int ncols) {
    int e = blockIdx.x;
    int s, d; edge_sd(e, src, dst, E, s, d);
    float norm = dinv[s] * dinv[d];
    for (int c = threadIdx.x; c < ncols; c += blockDim.x)
        atomicAdd(&out[(size_t)d * ldo + c], h[(size_t)s * ldh + c] * norm);
}

// ---------------- GAT pieces ----------------

__global__ void k_gat_scores(const float* __restrict__ h, int ldh, int colOff,
                             const float* __restrict__ as, const float* __restrict__ ad,
                             float* ssrc, float* sdst_s, int n, int heads, int C) {
    int i = blockIdx.x * blockDim.x + threadIdx.x;
    if (i >= n * heads) return;
    int node = i / heads, hd = i - node * heads;
    const float* hp  = h + (size_t)node * ldh + colOff + hd * C;
    const float* asp = as + hd * C;
    const float* adp = ad + hd * C;
    float s0 = 0.f, s1 = 0.f;
    for (int c = 0; c < C; c++) { float v = hp[c]; s0 += v * asp[c]; s1 += v * adp[c]; }
    ssrc[i] = s0; sdst_s[i] = s1;
}

__global__ void k_gat_logit_max(const float* __restrict__ ssrc, const float* __restrict__ sdst_s,
                                const int* __restrict__ src, const int* __restrict__ dst,
                                int E, int n, int heads, float* elog, float* m) {
    int i = blockIdx.x * blockDim.x + threadIdx.x;
    int total = (E + n) * heads;
    if (i >= total) return;
    int e = i / heads, hd = i - e * heads;
    int s, d; edge_sd(e, src, dst, E, s, d);
    float v = ssrc[s * heads + hd] + sdst_s[d * heads + hd];
    v = (v > 0.f) ? v : 0.2f * v;
    elog[i] = v;
    atomicMaxF(&m[d * heads + hd], v);
}

__global__ void k_gat_expsum(float* elog, const float* __restrict__ m,
                             const int* __restrict__ src, const int* __restrict__ dst,
                             int E, int n, int heads, float* ssum) {
    int i = blockIdx.x * blockDim.x + threadIdx.x;
    int total = (E + n) * heads;
    if (i >= total) return;
    int e = i / heads, hd = i - e * heads;
    int s, d; edge_sd(e, src, dst, E, s, d);
    float w = __expf(elog[i] - m[d * heads + hd]);
    elog[i] = w;
    atomicAdd(&ssum[d * heads + hd], w);
}

__global__ void k_gat_agg(const float* __restrict__ h, int ldh, int colOffIn,
                          const float* __restrict__ elog, const float* __restrict__ ssum,
                          const int* __restrict__ src, const int* __restrict__ dst,
                          int E, int n, int heads, int C,
                          float* out, int ldo, int colOffOut) {
    int e = blockIdx.x;
    int s, d; edge_sd(e, src, dst, E, s, d);
    int HC = heads * C;
    for (int c = threadIdx.x; c < HC; c += blockDim.x) {
        int hd = c / C;
        float alpha = elog[(size_t)e * heads + hd] / ssum[d * heads + hd];
        atomicAdd(&out[(size_t)d * ldo + colOffOut + c],
                  h[(size_t)s * ldh + colOffIn + c] * alpha);
    }
}

__global__ void k_lv_mean(const float* __restrict__ lvfull, const float* __restrict__ bias,
                          float* out, int n, int heads, int C) {
    int i = blockIdx.x * blockDim.x + threadIdx.x;
    if (i >= n * C) return;
    int node = i / C, c = i - node * C;
    float s = 0.f;
    for (int hd = 0; hd < heads; hd++) s += lvfull[(size_t)node * heads * C + hd * C + c];
    out[i] = s * (1.f / heads) + bias[c];
}

__global__ void k_z(const float* __restrict__ mu, const float* __restrict__ eps,
                    const float* __restrict__ lv, float* z, int n) {
    int i = blockIdx.x * blockDim.x + threadIdx.x;
    if (i < n) z[i] = mu[i] + eps[i] * __expf(lv[i]);
}

// ---------------- launcher ----------------

extern "C" void kernel_launch(void* const* d_in, const int* in_sizes, int n_in,
                              void* d_out, int out_size, void* d_ws, size_t ws_size,
                              hipStream_t stream) {
    const float* x      = (const float*)d_in[0];
    const int*   ei     = (const int*)d_in[1];
    const int*   src    = ei;
    const int*   dst    = ei + E_EDGES;
    const float* eps    = (const float*)d_in[2];
    const float* gcn0_W = (const float*)d_in[3];  const float* gcn0_b = (const float*)d_in[4];
    const float* gcn1_W = (const float*)d_in[5];  const float* gcn1_b = (const float*)d_in[6];
    const float* gat0_W = (const float*)d_in[7];  const float* gat0_as = (const float*)d_in[8];
    const float* gat0_ad = (const float*)d_in[9]; const float* gat0_b = (const float*)d_in[10];
    const float* gat1_W = (const float*)d_in[11]; const float* gat1_as = (const float*)d_in[12];
    const float* gat1_ad = (const float*)d_in[13];const float* gat1_b = (const float*)d_in[14];
    const float* mu_W   = (const float*)d_in[15]; const float* mu_b  = (const float*)d_in[16];
    const float* lv_W   = (const float*)d_in[17]; const float* lv_as = (const float*)d_in[18];
    const float* lv_ad  = (const float*)d_in[19]; const float* lv_b  = (const float*)d_in[20];

    float* adj    = (float*)d_out;
    float* mu_out = adj + (size_t)N_NODES * N_NODES;
    float* lv_out = mu_out + (size_t)N_NODES * H2D;

    float* ws = (float*)d_ws;
    size_t off = 0;
    auto alloc = [&](size_t nf) { float* p = ws + off; off += nf; return p; };
    float* dinv   = alloc(N_NODES);
    float* XW     = alloc((size_t)N_NODES * CATW);           // X @ [all W]  (pre-agg)
    float* Hc     = alloc((size_t)N_NODES * CATW);           // h_cat        (post-agg)
    float* ssrc   = alloc((size_t)N_NODES * HEADS);
    float* sdst_s = alloc((size_t)N_NODES * HEADS);
    float* elog   = alloc((size_t)(E_EDGES + N_NODES) * HEADS);
    float* segm   = alloc((size_t)N_NODES * HEADS);
    float* segs   = alloc((size_t)N_NODES * HEADS);
    float* XWmu   = alloc((size_t)N_NODES * H2D);
    float* XWlv   = alloc((size_t)N_NODES * HEADS * H2D);
    float* lvfull = alloc((size_t)N_NODES * HEADS * H2D);
    float* zbuf   = alloc((size_t)N_NODES * H2D);
    (void)ws_size; (void)in_sizes; (void)n_in; (void)out_size;

    const int EA = E_EDGES + N_NODES;

    auto gemm = [&](const float* A, int lda, const float* B, int ldb, int tB,
                    float* C, int ldc, const float* bias, int M, int N, int K, int sig) {
        dim3 grid((N + BN - 1) / BN, M / BM);
        k_gemm_tiled<<<grid, 256, 0, stream>>>(A, lda, B, ldb, tB, C, ldc, bias, M, N, K, sig);
    };
    auto fill = [&](float* p, float v, int n) {
        k_fill<<<(n + 255) / 256, 256, 0, stream>>>(p, v, n);
    };
    auto initb = [&](float* out, const float* b, int ncols, int ldo) {
        int total = N_NODES * ncols;
        k_init_bias<<<(total + 255) / 256, 256, 0, stream>>>(out, b, ncols, ldo, N_NODES);
    };

    // 1) degrees (self loops => start at 1), then dinv = rsqrt(deg)
    fill(dinv, 1.f, N_NODES);
    k_deg<<<(E_EDGES + 255) / 256, 256, 0, stream>>>(dst, dinv, E_EDGES);
    k_rsqrt<<<(N_NODES + 255) / 256, 256, 0, stream>>>(dinv, N_NODES);

    // 2) feature GEMMs: XW[:, slices] = x @ {gcn0_W, gcn1_W, gat0_W, gat1_W}
    gemm(x, F_DIM, gcn0_W, H1D,       0, XW + 0,   CATW, nullptr, N_NODES, H1D,       F_DIM, 0);
    gemm(x, F_DIM, gcn1_W, H1D,       0, XW + H1D, CATW, nullptr, N_NODES, H1D,       F_DIM, 0);
    gemm(x, F_DIM, gat0_W, HEADS*H1D, 0, XW + 256, CATW, nullptr, N_NODES, HEADS*H1D, F_DIM, 0);
    gemm(x, F_DIM, gat1_W, HEADS*H1D, 0, XW + 768, CATW, nullptr, N_NODES, HEADS*H1D, F_DIM, 0);

    // 3) GCN0 / GCN1 aggregation into h_cat slices [0:128), [128:256)
    initb(Hc + 0,   gcn0_b, H1D, CATW);
    k_gcn_agg<<<EA, 128, 0, stream>>>(XW + 0,   CATW, src, dst, E_EDGES, N_NODES, dinv, Hc + 0,   CATW, H1D);
    initb(Hc + H1D, gcn1_b, H1D, CATW);
    k_gcn_agg<<<EA, 128, 0, stream>>>(XW + H1D, CATW, src, dst, E_EDGES, N_NODES, dinv, Hc + H1D, CATW, H1D);

    // 4) GAT0 into h_cat [256:768)
    {
        int nh = N_NODES * HEADS, tot = EA * HEADS;
        k_gat_scores<<<(nh + 255) / 256, 256, 0, stream>>>(XW, CATW, 256, gat0_as, gat0_ad, ssrc, sdst_s, N_NODES, HEADS, H1D);
        fill(segm, -1e30f, nh); fill(segs, 0.f, nh);
        k_gat_logit_max<<<(tot + 255) / 256, 256, 0, stream>>>(ssrc, sdst_s, src, dst, E_EDGES, N_NODES, HEADS, elog, segm);
        k_gat_expsum<<<(tot + 255) / 256, 256, 0, stream>>>(elog, segm, src, dst, E_EDGES, N_NODES, HEADS, segs);
        initb(Hc + 256, gat0_b, HEADS * H1D, CATW);
        k_gat_agg<<<EA, 256, 0, stream>>>(XW, CATW, 256, elog, segs, src, dst, E_EDGES, N_NODES, HEADS, H1D, Hc, CATW, 256);
    }
    // 5) GAT1 into h_cat [768:1280)
    {
        int nh = N_NODES * HEADS, tot = EA * HEADS;
        k_gat_scores<<<(nh + 255) / 256, 256, 0, stream>>>(XW, CATW, 768, gat1_as, gat1_ad, ssrc, sdst_s, N_NODES, HEADS, H1D);
        fill(segm, -1e30f, nh); fill(segs, 0.f, nh);
        k_gat_logit_max<<<(tot + 255) / 256, 256, 0, stream>>>(ssrc, sdst_s, src, dst, E_EDGES, N_NODES, HEADS, elog, segm);
        k_gat_expsum<<<(tot + 255) / 256, 256, 0, stream>>>(elog, segm, src, dst, E_EDGES, N_NODES, HEADS, segs);
        initb(Hc + 768, gat1_b, HEADS * H1D, CATW);
        k_gat_agg<<<EA, 256, 0, stream>>>(XW, CATW, 768, elog, segs, src, dst, E_EDGES, N_NODES, HEADS, H1D, Hc, CATW, 768);
    }

    // 6) mu = GCN(h_cat; mu_W, mu_b)   -> d_out slice
    gemm(Hc, CATW, mu_W, H2D, 0, XWmu, H2D, nullptr, N_NODES, H2D, CATW, 0);
    initb(mu_out, mu_b, H2D, H2D);
    k_gcn_agg<<<EA, 64, 0, stream>>>(XWmu, H2D, src, dst, E_EDGES, N_NODES, dinv, mu_out, H2D, H2D);

    // 7) logvar = GAT(h_cat; lv_*, concat=False)  -> d_out slice
    {
        int nh = N_NODES * HEADS, tot = EA * HEADS;
        gemm(Hc, CATW, lv_W, HEADS * H2D, 0, XWlv, HEADS * H2D, nullptr, N_NODES, HEADS * H2D, CATW, 0);
        k_gat_scores<<<(nh + 255) / 256, 256, 0, stream>>>(XWlv, HEADS * H2D, 0, lv_as, lv_ad, ssrc, sdst_s, N_NODES, HEADS, H2D);
        fill(segm, -1e30f, nh); fill(segs, 0.f, nh);
        k_gat_logit_max<<<(tot + 255) / 256, 256, 0, stream>>>(ssrc, sdst_s, src, dst, E_EDGES, N_NODES, HEADS, elog, segm);
        k_gat_expsum<<<(tot + 255) / 256, 256, 0, stream>>>(elog, segm, src, dst, E_EDGES, N_NODES, HEADS, segs);
        fill(lvfull, 0.f, N_NODES * HEADS * H2D);
        k_gat_agg<<<EA, 256, 0, stream>>>(XWlv, HEADS * H2D, 0, elog, segs, src, dst, E_EDGES, N_NODES, HEADS, H2D, lvfull, HEADS * H2D, 0);
        int nc = N_NODES * H2D;
        k_lv_mean<<<(nc + 255) / 256, 256, 0, stream>>>(lvfull, lv_b, lv_out, N_NODES, HEADS, H2D);
    }

    // 8) z = mu + eps * exp(logvar)
    {
        int nz = N_NODES * H2D;
        k_z<<<(nz + 255) / 256, 256, 0, stream>>>(mu_out, eps, lv_out, zbuf, nz);
    }

    // 9) adj = sigmoid(z @ z.T)
    gemm(zbuf, H2D, zbuf, H2D, /*transB=*/1, adj, N_NODES, nullptr, N_NODES, N_NODES, H2D, /*sigmoid=*/1);
}
